// MultiheadAttentionBias_88837103551299
// MI455X (gfx1250) — matrix operand order repaired and checked by reading, not compiled, not run on hardware
//
#include <hip/hip_runtime.h>

// Problem constants (from reference): L=S=1024, N=8, E=1024, H=16, hd=64, B=N*H=128
#define L_DIM   1024
#define S_DIM   1024
#define N_BATCH 8
#define E_DIM   1024
#define H_HEADS 16
#define HD      64
#define BH      128

typedef __attribute__((ext_vector_type(16))) _Float16 v16h;
typedef __attribute__((ext_vector_type(8)))  _Float16 v8h;
typedef __attribute__((ext_vector_type(8)))  float    v8f;
typedef __attribute__((ext_vector_type(4)))  int      v4i;

#if __has_builtin(__builtin_amdgcn_global_load_async_to_lds_b128) && \
    __has_builtin(__builtin_amdgcn_s_wait_asynccnt)
#define USE_ASYNC_LDS 1
// Builtin signature (from hipcc diagnostic): (v4i AS1*, v4i AS3*, imm offset, imm cpol)
typedef __attribute__((address_space(1))) v4i as1_v4i;
typedef __attribute__((address_space(3))) v4i as3_v4i;
#endif

// ---- fragment builders -----------------------------------------------------
// A-matrix 16x32 f16 layout: lane l<16 holds row M=l with
// K = ka..ka+15; lane l+16 holds same row with ka += 16.
__device__ __forceinline__ v16h a_frag_from_f32(const float* __restrict__ row, int ka) {
  float4 x0 = *(const float4*)(row + ka);
  float4 x1 = *(const float4*)(row + ka + 4);
  float4 y0 = *(const float4*)(row + ka + 8);
  float4 y1 = *(const float4*)(row + ka + 12);
  v16h a;
  a[0]  = (_Float16)x0.x; a[1]  = (_Float16)x0.y; a[2]  = (_Float16)x0.z; a[3]  = (_Float16)x0.w;
  a[4]  = (_Float16)x1.x; a[5]  = (_Float16)x1.y; a[6]  = (_Float16)x1.z; a[7]  = (_Float16)x1.w;
  a[8]  = (_Float16)y0.x; a[9]  = (_Float16)y0.y; a[10] = (_Float16)y0.z; a[11] = (_Float16)y0.w;
  a[12] = (_Float16)y1.x; a[13] = (_Float16)y1.y; a[14] = (_Float16)y1.z; a[15] = (_Float16)y1.w;
  return a;
}

__device__ __forceinline__ v16h a_frag_from_f16(const _Float16* __restrict__ row, int ka) {
  v8h lo = *(const v8h*)(row + ka);
  v8h hi = *(const v8h*)(row + ka + 8);
  v16h a;
#pragma unroll
  for (int i = 0; i < 8; ++i) { a[i] = lo[i]; a[8 + i] = hi[i]; }
  return a;
}

// B-matrix 32x16 f16 layout as used here (same k order as A): lane n<16 holds
// column N=n with K = 0..15, lane n+16 holds K = 16..31, packed 2 per VGPR.
__device__ __forceinline__ v16h b_frag_from_f32(const float* __restrict__ p) {
  float4 f0 = *(const float4*)(p + 0);
  float4 f1 = *(const float4*)(p + 4);
  float4 f2 = *(const float4*)(p + 8);
  float4 f3 = *(const float4*)(p + 12);
  v16h b;
  b[0]  = (_Float16)f0.x; b[1]  = (_Float16)f0.y; b[2]  = (_Float16)f0.z; b[3]  = (_Float16)f0.w;
  b[4]  = (_Float16)f1.x; b[5]  = (_Float16)f1.y; b[6]  = (_Float16)f1.z; b[7]  = (_Float16)f1.w;
  b[8]  = (_Float16)f2.x; b[9]  = (_Float16)f2.y; b[10] = (_Float16)f2.z; b[11] = (_Float16)f2.w;
  b[12] = (_Float16)f3.x; b[13] = (_Float16)f3.y; b[14] = (_Float16)f3.z; b[15] = (_Float16)f3.w;
  return b;
}

#define WMMA_F16(a, b, c) \
  __builtin_amdgcn_wmma_f32_16x16x32_f16(false, (a), false, (b), (short)0, (c), false, false)

// ---- Kernel 1: Q/K projection GEMM, f32 -> f16 head layout -----------------
// z==0: Q = (query @ Wq^T + bq) * hd^-0.5 ; z==1: K = key @ Wk^T + bk
__global__ void __launch_bounds__(128)
mha_proj_kernel(const float* __restrict__ query, const float* __restrict__ key,
                const float* __restrict__ w, const float* __restrict__ bias,
                _Float16* __restrict__ qh, _Float16* __restrict__ kh) {
  const int lane = threadIdx.x & 31;
  const int wave = threadIdx.x >> 5;
  const int half = lane >> 4;
  const int r16  = lane & 15;

  const bool is_q = (blockIdx.z == 0);
  const float* A  = is_q ? query : key;                      // (8192, 1024), m = l*N + n
  const float* W  = w + (is_q ? 0 : (size_t)E_DIM * E_DIM);  // rows = out features
  const float* bv = bias + (is_q ? 0 : E_DIM);
  _Float16* out   = is_q ? qh : kh;

  const int m0 = blockIdx.x * 64 + wave * 16;
  const int n0 = blockIdx.y * 64;

  const float* rowA = A + (size_t)(m0 + r16) * E_DIM;

  v8f acc[4] = {v8f{}, v8f{}, v8f{}, v8f{}};
  for (int kb = 0; kb < E_DIM; kb += 32) {
    v16h a = a_frag_from_f32(rowA, kb + half * 16);
    const int kB = kb + half * 16;
#pragma unroll
    for (int t = 0; t < 4; ++t) {
      v16h b = b_frag_from_f32(W + (size_t)(n0 + t * 16 + r16) * E_DIM + kB);
      acc[t] = WMMA_F16(a, b, acc[t]);
    }
  }

  const float scale = is_q ? 0.125f : 1.0f;
#pragma unroll
  for (int t = 0; t < 4; ++t) {
    const int col  = n0 + t * 16 + r16;
    const int h    = col >> 6;
    const int d    = col & (HD - 1);
    const float bc = bv[col];
#pragma unroll
    for (int r = 0; r < 8; ++r) {
      const int row = m0 + half * 8 + r;  // C layout: lanes 0-15 -> M=r, 16-31 -> M=8+r
      const int l   = row >> 3;
      const int nb  = row & (N_BATCH - 1);
      const float v = (acc[t][r] + bc) * scale;
      out[(((size_t)(nb * H_HEADS + h)) * L_DIM + l) * HD + d] = (_Float16)v;
    }
  }
}

// ---- Kernel 2: scores = qh @ kh^T + bias, row softmax ----------------------
// One wave per (head b, 16-row block l0). 64 KB LDS score tile. The bias tile
// for (b, l0..l0+15, :) is a contiguous 64 KB region: async-DMA it into LDS
// (ASYNCcnt path), overlapping with Q-fragment loads, and fold
// the bias add into the score store.
__global__ void __launch_bounds__(32)
mha_score_softmax_kernel(const _Float16* __restrict__ qh, const _Float16* __restrict__ kh,
                         const float* __restrict__ attn_bias, float* __restrict__ out) {
  __shared__ float sm[16 * S_DIM];  // 64 KB (WGP has 320 KB)

  const int lane = threadIdx.x & 31;
  const int half = lane >> 4;
  const int r16  = lane & 15;
  const int b    = blockIdx.y;
  const int l0   = blockIdx.x * 16;

  const float* bias_base = attn_bias + ((size_t)b * L_DIM + l0) * S_DIM;

#ifdef USE_ASYNC_LDS
  // 64 KB contiguous global -> LDS, 16 B per lane per op, 128 ops (self-throttled
  // by ASYNCcnt). Runs concurrently with the fragment loads below.
  {
    const char* g = (const char*)bias_base;
    char* l = (char*)sm;
    for (int i = 0; i < 128; ++i) {
      const int off = (i * 32 + lane) * 16;
      __builtin_amdgcn_global_load_async_to_lds_b128(
          (as1_v4i*)(g + off), (as3_v4i*)(l + off), 0, 0);
    }
  }
#else
#pragma unroll
  for (int r = 0; r < 16; ++r)
    __builtin_prefetch(bias_base + (size_t)r * S_DIM + lane * 32, 0, 1);
#endif

  // Q fragments (reused for all 64 S-tiles): contraction d = 0..63, two K=32 steps
  const _Float16* qrow = qh + ((size_t)b * L_DIM + (l0 + r16)) * HD;
  v16h a0 = a_frag_from_f16(qrow, 0 + half * 16);
  v16h a1 = a_frag_from_f16(qrow, 32 + half * 16);

#ifdef USE_ASYNC_LDS
  __builtin_amdgcn_s_wait_asynccnt(0);
#endif
  __syncthreads();

  for (int st = 0; st < S_DIM / 16; ++st) {
    const int s0 = st * 16;
    // B[d][n] = kh[b][s0+n][d]: lane n holds 16 contiguous d-values -> one 32B load
    const _Float16* krow = kh + ((size_t)b * S_DIM + (s0 + r16)) * HD;
    v16h b0 = *(const v16h*)(krow + half * 16);
    v16h b1 = *(const v16h*)(krow + 32 + half * 16);
    v8f c = v8f{};
    c = WMMA_F16(a0, b0, c);
    c = WMMA_F16(a1, b1, c);
#pragma unroll
    for (int r = 0; r < 8; ++r) {
      const int idx = (half * 8 + r) * S_DIM + s0 + r16;
#ifdef USE_ASYNC_LDS
      sm[idx] += c[r];   // bias already resident in LDS
#else
      sm[idx] = c[r];
#endif
    }
  }
  __syncthreads();

#ifndef USE_ASYNC_LDS
  // Fallback: coalesced b128 bias add pass
  for (int r = 0; r < 16; ++r) {
    const float4* brow = (const float4*)(bias_base + (size_t)r * S_DIM);
    float4* srow = (float4*)(sm + r * S_DIM);
    for (int c4 = lane; c4 < S_DIM / 4; c4 += 32) {
      float4 bv = brow[c4];
      float4 sv = srow[c4];
      sv.x += bv.x; sv.y += bv.y; sv.z += bv.z; sv.w += bv.w;
      srow[c4] = sv;
    }
  }
  __syncthreads();
#endif

  // Numerically-stable softmax per row: b128 LDS reads, b128 global writes
  for (int r = 0; r < 16; ++r) {
    const float4* srow4 = (const float4*)(sm + r * S_DIM);
    float m = -3.402823466e38f;
    for (int c4 = lane; c4 < S_DIM / 4; c4 += 32) {
      float4 v = srow4[c4];
      m = fmaxf(m, fmaxf(fmaxf(v.x, v.y), fmaxf(v.z, v.w)));
    }
#pragma unroll
    for (int off = 16; off > 0; off >>= 1) m = fmaxf(m, __shfl_xor(m, off, 32));

    float s = 0.0f;
    for (int c4 = lane; c4 < S_DIM / 4; c4 += 32) {
      float4 v = srow4[c4];
      s += __expf(v.x - m) + __expf(v.y - m) + __expf(v.z - m) + __expf(v.w - m);
    }
#pragma unroll
    for (int off = 16; off > 0; off >>= 1) s += __shfl_xor(s, off, 32);

    const float inv = 1.0f / s;
    float4* orow4 = (float4*)(out + ((size_t)b * L_DIM + (l0 + r)) * S_DIM);
    for (int c4 = lane; c4 < S_DIM / 4; c4 += 32) {
      float4 v = srow4[c4];
      float4 o;
      o.x = __expf(v.x - m) * inv;
      o.y = __expf(v.y - m) * inv;
      o.z = __expf(v.z - m) * inv;
      o.w = __expf(v.w - m) * inv;
      orow4[c4] = o;
    }
  }
}

extern "C" void kernel_launch(void* const* d_in, const int* in_sizes, int n_in,
                              void* d_out, int out_size, void* d_ws, size_t ws_size,
                              hipStream_t stream) {
  const float* query     = (const float*)d_in[0];  // (L, N, E)
  const float* key       = (const float*)d_in[1];  // (S, N, E)
  const float* w         = (const float*)d_in[2];  // (3E, E)
  const float* bias      = (const float*)d_in[3];  // (3E,)
  const float* attn_bias = (const float*)d_in[4];  // (B, L, S)
  float* out = (float*)d_out;                      // (B, L, S)

  _Float16* qh = (_Float16*)d_ws;                  // 16 MB
  _Float16* kh = qh + (size_t)BH * L_DIM * HD;     // 16 MB

  dim3 g1((L_DIM * N_BATCH) / 64, E_DIM / 64, 2);
  mha_proj_kernel<<<g1, 128, 0, stream>>>(query, key, w, bias, qh, kh);

  dim3 g2(L_DIM / 16, BH);
  mha_score_softmax_kernel<<<g2, 32, 0, stream>>>(qh, kh, attn_bias, out);
}
